// GNNEncoder_30537217474822
// MI455X (gfx1250) — matrix operand order repaired and checked by reading, not compiled, not run on hardware
//
#include <hip/hip_runtime.h>
#include <hip/hip_bf16.h>

// ---------- types ----------
typedef __attribute__((ext_vector_type(16))) __bf16 v16bf;
typedef __attribute__((ext_vector_type(8)))  __bf16 v8bf;
typedef __attribute__((ext_vector_type(8)))  float  v8f;

// ---------- problem constants ----------
#define N_NODES 200000
#define N_EDGES 3200000
#define DIM     514
#define NB      16
#define OUTD    256
#define KP      544     // K padded to 17*32
#define LDH     528     // output cols padded to 33*16
#define EPSV    1e-5f

// ---------- degrees ----------
__global__ void deg_kernel(const int* __restrict__ src, const int* __restrict__ dst,
                           float* __restrict__ dout, float* __restrict__ din) {
    int e = blockIdx.x * blockDim.x + threadIdx.x;
    if (e >= N_EDGES) return;
    atomicAdd(&dout[src[e]], 1.0f);
    atomicAdd(&din[dst[e]], 1.0f);
}

__global__ void cnt_kernel(const int* __restrict__ gid, float* __restrict__ cnt) {
    int n = blockIdx.x * blockDim.x + threadIdx.x;
    if (n >= N_NODES) return;
    atomicAdd(&cnt[gid[n]], 1.0f);
}

__global__ void rdeg_kernel(float* __restrict__ dout, float* __restrict__ din) {
    int n = blockIdx.x * blockDim.x + threadIdx.x;
    if (n >= N_NODES) return;
    dout[n] = rsqrtf(fmaxf(dout[n], 1.0f));
    din[n]  = rsqrtf(fmaxf(din[n], 1.0f));
}

// ---------- edge gather-scatter: agg[dst] += x[src] * rsqrt(deg_out[src]) ----------
// one wave32 per edge; lanes stride the 514 features -> coalesced atomics
__global__ void scatter_kernel(const float* __restrict__ x,
                               const int* __restrict__ src, const int* __restrict__ dst,
                               const float* __restrict__ rdout, float* __restrict__ agg) {
    int lane = threadIdx.x & 31;
    int e = blockIdx.x * 8 + (threadIdx.x >> 5);
    if (e >= N_EDGES) return;
    int s = src[e], d = dst[e];
    float w = rdout[s];
    const float* xs = x + (size_t)s * DIM;
    float* ad = agg + (size_t)d * DIM;
    for (int k = lane; k < DIM; k += 32)
        atomicAdd(&ad[k], xs[k] * w);
}

// ---------- pack A: bf16, scaled by rsqrt(deg_in), K zero-padded to 544 ----------
__global__ void packA_kernel(const float* __restrict__ agg, const float* __restrict__ rdin,
                             __bf16* __restrict__ A) {
    int n = blockIdx.x;
    float w = rdin[n];
    const float* ar = agg + (size_t)n * DIM;
    __bf16* Ar = A + (size_t)n * KP;
    for (int k = threadIdx.x; k < KP; k += blockDim.x)
        Ar[k] = (k < DIM) ? (__bf16)(ar[k] * w) : (__bf16)0.0f;
}

// ---------- pack W transposed: Wt[j][k] = W[k][j], bf16, zero-padded ----------
__global__ void packW_kernel(const float* __restrict__ W, __bf16* __restrict__ Wt) {
    int j = blockIdx.x;                 // 0..527
    __bf16* r = Wt + (size_t)j * KP;
    for (int k = threadIdx.x; k < KP; k += blockDim.x)
        r[k] = (j < DIM && k < DIM) ? (__bf16)W[(size_t)k * DIM + j] : (__bf16)0.0f;
}

// ---------- GEMM: h[N,528] = A(bf16)[N,544] x Wt^T + bias ----------
// Templated on NT (16-col tiles per wave) so the k-loop is branch-free.
// NT=8 covers cols [jbase0 + y*128, +128); NT=1 handles the 512..513 fringe.
template <int NT>
__global__ void __launch_bounds__(256) gemm_kernel_t(const __bf16* __restrict__ A,
                                                     const __bf16* __restrict__ Wt,
                                                     const float* __restrict__ bias,
                                                     float* __restrict__ h,
                                                     int jbase0) {
    constexpr int CH = (NT < 4) ? NT : 4;            // B-fragment group size (reg cap)
    int lane  = threadIdx.x & 31;
    int wave  = threadIdx.x >> 5;
    int mtile = blockIdx.x * 8 + wave;               // wave-uniform
    if (mtile >= N_NODES / 16) return;               // EXEC all-ones for live waves
    int row_base = mtile * 16;
    int jbase = jbase0 + blockIdx.y * (NT * 16);

    int lm = lane & 15;                              // M (A) / N (B,C) within tile
    int hi = lane >> 4;

    // A: lanes 0-15 hold K = {0..15}, lanes 16-31 hold K = {16..31}
    const __bf16* Arow  = A  + (size_t)(row_base + lm) * KP + (hi << 4);
    // B: lane holds Wt[j][k0 + (hi?16:0) .. +15] contiguous (32B load);
    // jt tiles are at compile-time byte offsets jt*16*KP*2 (fits 24-bit IOFFSET)
    const __bf16* Bbase = Wt + (size_t)(jbase + lm) * KP + (hi << 4);

    v8f acc[NT];
#pragma unroll
    for (int jt = 0; jt < NT; ++jt) {
        int j = jbase + jt * 16 + lm;
        float bj = (j < DIM) ? bias[j] : 0.0f;
        acc[jt] = (v8f){bj, bj, bj, bj, bj, bj, bj, bj};
    }

    for (int k0 = 0; k0 < KP; k0 += 32) {
        v8bf alo = *(const v8bf*)(Arow + k0);
        v8bf ahi = *(const v8bf*)(Arow + k0 + 8);
        v16bf af = __builtin_shufflevector(alo, ahi,
                                           0,1,2,3,4,5,6,7,8,9,10,11,12,13,14,15);
#pragma unroll
        for (int g0 = 0; g0 < NT; g0 += CH) {
            v16bf bf[CH];
#pragma unroll
            for (int u = 0; u < CH; ++u)
                bf[u] = *(const v16bf*)(Bbase + (g0 + u) * (16 * KP) + k0);
#pragma unroll
            for (int u = 0; u < CH; ++u)
                acc[g0 + u] = __builtin_amdgcn_wmma_f32_16x16x32_bf16(
                    false, af, false, bf[u], (short)0, acc[g0 + u], false, false);
        }
    }

#pragma unroll
    for (int jt = 0; jt < NT; ++jt) {
        int j = jbase + jt * 16 + lm;
        if (j < DIM) {
#pragma unroll
            for (int r = 0; r < 8; ++r) {
                int row = row_base + r + (hi << 3);  // VGPR r: M=r (lanes<16) / r+8
                h[(size_t)row * LDH + j] = acc[jt][r];
            }
        }
    }
}

// ---------- fused column stats + per-graph sums over h ----------
__global__ void stats_kernel(const float* __restrict__ h, const int* __restrict__ gid,
                             float* __restrict__ colsum, float* __restrict__ colsum2,
                             float* __restrict__ gsum) {
    __shared__ int sg[256];
    int r0 = blockIdx.x * 256;
    int rcount = N_NODES - r0; if (rcount > 256) rcount = 256;
    int tid = threadIdx.x;
    if (tid < rcount) sg[tid] = gid[r0 + tid];
    __syncthreads();
    int c = blockIdx.y * 256 + tid;
    if (c < DIM) {
        float s1 = 0.f, s2 = 0.f, gacc = 0.f;
        int curg = sg[0];
        for (int i = 0; i < rcount; ++i) {
            float v = h[(size_t)(r0 + i) * LDH + c];
            s1 += v; s2 += v * v;
            int g = sg[i];
            if (g != curg) {                       // graph_ids sorted -> rare flushes
                atomicAdd(&gsum[curg * KP + c], gacc);
                gacc = 0.f; curg = g;
            }
            gacc += v;
        }
        atomicAdd(&gsum[curg * KP + c], gacc);
        atomicAdd(&colsum[c], s1);
        atomicAdd(&colsum2[c], s2);
    }
}

// ---------- collapse GraphNorm+BatchNorm to out = s*h + t ----------
__global__ void finalize_kernel(const float* __restrict__ colsum, const float* __restrict__ colsum2,
                                const float* __restrict__ gn_alpha, const float* __restrict__ gn_w,
                                const float* __restrict__ bn_g, const float* __restrict__ bn_b,
                                float* __restrict__ sbuf, float* __restrict__ tbuf) {
    int c = threadIdx.x;
    if (c >= DIM) return;
    float invN = 1.0f / (float)N_NODES;
    float m1 = colsum[c] * invN;
    float m2 = colsum2[c] * invN;
    float a  = gn_alpha[c];
    float varg = m2 - (2.0f * a - a * a) * m1 * m1;   // E[(h - a*m1)^2]
    float g = gn_w[c] * rsqrtf(varg + EPSV);
    float v = g * g * (m2 - m1 * m1);                 // BN variance of g*(h-m1)+const
    float s = bn_g[c] * g * rsqrtf(v + EPSV);
    sbuf[c] = s;
    tbuf[c] = bn_b[c] - s * m1;                       // gn_bias cancels in BN
}

// ---------- pooled = s*(gsum/cnt)+t ; out = pooled @ lin_W + lin_b ----------
__global__ void out_kernel(const float* __restrict__ gsum, const float* __restrict__ cnt,
                           const float* __restrict__ sbuf, const float* __restrict__ tbuf,
                           const float* __restrict__ linW, const float* __restrict__ linb,
                           float* __restrict__ out) {
    __shared__ float pooled[DIM];
    int b = blockIdx.x;
    float rc = 1.0f / fmaxf(cnt[b], 1.0f);
    for (int d = threadIdx.x; d < DIM; d += blockDim.x)
        pooled[d] = sbuf[d] * (gsum[b * KP + d] * rc) + tbuf[d];
    __syncthreads();
    int o = threadIdx.x;
    float acc = linb[o];
    for (int d = 0; d < DIM; ++d)
        acc += pooled[d] * linW[(size_t)d * OUTD + o];
    out[b * OUTD + o] = acc;
}

// ---------- launch ----------
extern "C" void kernel_launch(void* const* d_in, const int* in_sizes, int n_in,
                              void* d_out, int out_size, void* d_ws, size_t ws_size,
                              hipStream_t stream) {
    const float* x        = (const float*)d_in[0];
    const float* W        = (const float*)d_in[1];
    const float* bias     = (const float*)d_in[2];
    const float* gn_w     = (const float*)d_in[3];
    /* d_in[4] = gn_bias: provably cancels through BatchNorm */
    const float* gn_alpha = (const float*)d_in[5];
    const float* bn_g     = (const float*)d_in[6];
    const float* bn_b     = (const float*)d_in[7];
    const float* linW     = (const float*)d_in[8];
    const float* linb     = (const float*)d_in[9];
    const int*   ei       = (const int*)d_in[10];
    const int*   src      = ei;
    const int*   dst      = ei + N_EDGES;
    const int*   gid      = (const int*)d_in[11];

    char* ws = (char*)d_ws;
    size_t off = 0;
    auto alloc = [&](size_t bytes) -> char* {
        char* p = ws + off;
        off += (bytes + 255) & ~(size_t)255;
        return p;
    };
    float*  deg_out = (float*)alloc((size_t)N_NODES * 4);
    float*  deg_in  = (float*)alloc((size_t)N_NODES * 4);
    float*  agg     = (float*)alloc((size_t)N_NODES * DIM * 4);
    __bf16* Abf     = (__bf16*)alloc((size_t)N_NODES * KP * 2);
    __bf16* Wt      = (__bf16*)alloc((size_t)LDH * KP * 2);
    float*  h       = (float*)alloc((size_t)N_NODES * LDH * 4);
    float*  colsum  = (float*)alloc(KP * 4);
    float*  colsum2 = (float*)alloc(KP * 4);
    float*  sbuf    = (float*)alloc(KP * 4);
    float*  tbuf    = (float*)alloc(KP * 4);
    float*  gsum    = (float*)alloc((size_t)NB * KP * 4);
    float*  cnt     = (float*)alloc((size_t)NB * 4);

    hipMemsetAsync(deg_out, 0, (size_t)N_NODES * 4, stream);
    hipMemsetAsync(deg_in,  0, (size_t)N_NODES * 4, stream);
    hipMemsetAsync(agg,     0, (size_t)N_NODES * DIM * 4, stream);
    hipMemsetAsync(colsum,  0, KP * 4, stream);
    hipMemsetAsync(colsum2, 0, KP * 4, stream);
    hipMemsetAsync(gsum,    0, (size_t)NB * KP * 4, stream);
    hipMemsetAsync(cnt,     0, (size_t)NB * 4, stream);

    deg_kernel<<<(N_EDGES + 255) / 256, 256, 0, stream>>>(src, dst, deg_out, deg_in);
    cnt_kernel<<<(N_NODES + 255) / 256, 256, 0, stream>>>(gid, cnt);
    rdeg_kernel<<<(N_NODES + 255) / 256, 256, 0, stream>>>(deg_out, deg_in);
    scatter_kernel<<<(N_EDGES + 7) / 8, 256, 0, stream>>>(x, src, dst, deg_out, agg);
    packA_kernel<<<N_NODES, 256, 0, stream>>>(agg, deg_in, Abf);
    packW_kernel<<<LDH, 256, 0, stream>>>(W, Wt);

    dim3 ggrid((N_NODES / 16 + 7) / 8, 4);           // cols 0..511, branch-free NT=8
    gemm_kernel_t<8><<<ggrid, 256, 0, stream>>>(Abf, Wt, bias, h, 0);
    dim3 fgrid((N_NODES / 16 + 7) / 8, 1);           // cols 512..513 fringe, NT=1
    gemm_kernel_t<1><<<fgrid, 256, 0, stream>>>(Abf, Wt, bias, h, 512);

    stats_kernel<<<dim3((N_NODES + 255) / 256, 3), 256, 0, stream>>>(h, gid, colsum, colsum2, gsum);
    finalize_kernel<<<1, KP, 0, stream>>>(colsum, colsum2, gn_alpha, gn_w, bn_g, bn_b, sbuf, tbuf);
    out_kernel<<<NB, 256, 0, stream>>>(gsum, cnt, sbuf, tbuf, linW, linb, (float*)d_out);
}
